// ParallelSMLSTMCell_80736795230387
// MI455X (gfx1250) — matrix operand order repaired and checked by reading, not compiled, not run on hardware
//
#include <hip/hip_runtime.h>
#include <hip/hip_bf16.h>

typedef __attribute__((ext_vector_type(16))) _Float16 v16h;
typedef __attribute__((ext_vector_type(8)))  float    v8f;

#define B_   8
#define S_   512
#define IN_  128
#define H_   64
#define DS_  128
#define D_   4096
#define WIN  102        // max(1, int(512*0.2))
#define NCOL 336        // q(64)|k(64)|v(64)|o(64)|r1(64)|i|f|pad->336
#define SC   32         // chunk length for linear-attention formulation

// ---------------- generic helpers ----------------

__device__ __forceinline__ float blkReduce(float v, float* red, int tid, int nthr) {
  red[tid] = v; __syncthreads();
  for (int off = nthr >> 1; off > 0; off >>= 1) {
    if (tid < off) red[tid] += red[tid + off];
    __syncthreads();
  }
  float r = red[0]; __syncthreads();
  return r;
}

// A-matrix 16x32 f16 loader (row-major buffer, row stride ld).
// lanes 0-15 hold K=0..15 ; lanes 16-31 hold K=16..31 (contiguous, matches B loaders)
__device__ __forceinline__ v16h ldA(const _Float16* buf, int ld, int row0, int k0, int lane) {
  int m = lane & 15, hi = lane >> 4;
  const _Float16* p = buf + (size_t)(row0 + m) * ld + k0 + 16 * hi;
  v16h a;
#pragma unroll
  for (int j = 0; j < 16; ++j) a[j] = p[j];
  return a;
}

// B-matrix 32x16 f16 loader from N-major storage bufT[n][k] (row stride ld).
// lanes 0-15: K=0..15, lanes 16-31: K=16..31 (contiguous in k).
__device__ __forceinline__ v16h ldBT(const _Float16* bufT, int ld, int k0, int n0, int lane) {
  int n = lane & 15, hi = lane >> 4;
  const _Float16* p = bufT + (size_t)(n0 + n) * ld + k0 + 16 * hi;
  v16h b;
#pragma unroll
  for (int j = 0; j < 16; ++j) b[j] = p[j];
  return b;
}

// B-matrix 32x16 loader from f32 buffer with general strides: elem(k,n)=buf[k*sK+n*sN]
__device__ __forceinline__ v16h ldBf32(const float* buf, int sK, int sN, int k0, int n0, int lane) {
  int n = lane & 15, hi = lane >> 4;
  const float* p = buf + (size_t)(k0 + 16 * hi) * sK + (size_t)(n0 + n) * sN;
  v16h b;
#pragma unroll
  for (int j = 0; j < 16; ++j) b[j] = (_Float16)p[j * sK];
  return b;
}

// D/C 16x16 f32: lane -> n=lane&15, elem i -> m=i+8*(lane>>4)
__device__ __forceinline__ void stD(float* buf, int ld, int m0, int n0, int lane, v8f c) {
  int n = lane & 15, hi = lane >> 4;
#pragma unroll
  for (int i = 0; i < 8; ++i) buf[(size_t)(m0 + i + 8 * hi) * ld + n0 + n] = c[i];
}
__device__ __forceinline__ v8f ldC(const float* buf, int ld, int m0, int n0, int lane) {
  int n = lane & 15, hi = lane >> 4;
  v8f c;
#pragma unroll
  for (int i = 0; i < 8; ++i) c[i] = buf[(size_t)(m0 + i + 8 * hi) * ld + n0 + n];
  return c;
}

__device__ __forceinline__ v8f wmma_f16(v16h a, v16h b, v8f c) {
  return __builtin_amdgcn_wmma_f32_16x16x32_f16(false, a, false, b, (short)0, c, false, false);
}

// ---------------- K0: pack all projection weights into one f16 N-major matrix ----------------
__global__ void k_prep(const float* Wq, const float* bq, const float* Wk, const float* bk,
                       const float* Wv, const float* bv, const float* Wi, const float* bi,
                       const float* Wf, const float* bf, const float* Wo, const float* bo,
                       const float* Wr1, const float* br1,
                       _Float16* WcatT, float* biascat) {
  int n = blockIdx.x;   // 0..335
  int k = threadIdx.x;  // 0..127
  float w = 0.f, bb = 0.f;
  if      (n < 64)  { w = Wq [(size_t)n * IN_ + k];        bb = bq[n]; }
  else if (n < 128) { w = Wk [(size_t)(n-64) * IN_ + k];   bb = bk[n-64]; }
  else if (n < 192) { w = Wv [(size_t)(n-128) * IN_ + k];  bb = bv[n-128]; }
  else if (n < 256) { w = Wo [(size_t)(n-192) * IN_ + k];  bb = bo[n-192]; }
  else if (n < 320) { w = Wr1[(size_t)(n-256) * IN_ + k];  bb = br1[n-256]; }
  else if (n == 320){ w = Wi[k];                           bb = bi[0]; }
  else if (n == 321){ w = Wf[k];                           bb = bf[0]; }
  WcatT[(size_t)n * IN_ + k] = (_Float16)w;
  if (k == 0) biascat[n] = bb;
}

// ---------------- K1: prefix sums over S per (b,f) for windowed stats ----------------
__global__ void k_cumsum(const float* x, float* cs, float* cs2) {
  int b = blockIdx.x, f = threadIdx.x;
  size_t base = (size_t)b * (S_ + 1) * IN_;
  float s = 0.f, s2 = 0.f;
  cs[base + f] = 0.f; cs2[base + f] = 0.f;
  for (int t = 0; t < S_; ++t) {
    float v = x[((size_t)b * S_ + t) * IN_ + f];
    s += v; s2 += v * v;
    cs [base + (size_t)(t + 1) * IN_ + f] = s;
    cs2[base + (size_t)(t + 1) * IN_ + f] = s2;
  }
}

// ---------------- K2: windowed std -> raw volatility ----------------
__global__ void k_vol(const float* cs, const float* cs2, float* volraw) {
  int bs = blockIdx.x; int b = bs / S_, t = bs % S_, f = threadIdx.x;
  __shared__ float red[128];
  int start = (t < WIN - 1) ? 0 : t - WIN + 1;
  int end = start + WIN;
  size_t base = (size_t)b * (S_ + 1) * IN_;
  float su = cs [base + (size_t)end * IN_ + f] - cs [base + (size_t)start * IN_ + f];
  float sq = cs2[base + (size_t)end * IN_ + f] - cs2[base + (size_t)start * IN_ + f];
  float mean = su / (float)WIN;
  float var = fmaxf((sq - WIN * mean * mean) / (float)(WIN - 1), 0.f);
  float tot = blkReduce(sqrtf(var), red, f, 128);
  if (f == 0) volraw[bs] = tot / 128.f;
}

// ---------------- K3: normalize vol by its per-batch mean over S ----------------
__global__ void k_volnorm(const float* volraw, float* vol) {
  int b = blockIdx.x, tid = threadIdx.x;
  __shared__ float red[256];
  float s = 0.f;
  for (int t = tid; t < S_; t += 256) s += volraw[b * S_ + t];
  red[tid] = s; __syncthreads();
  for (int off = 128; off > 0; off >>= 1) { if (tid < off) red[tid] += red[tid + off]; __syncthreads(); }
  float m = red[0] / (float)S_;
  for (int t = tid; t < S_; t += 256) vol[b * S_ + t] = volraw[b * S_ + t] / (m + 1e-6f);
}

// ---------------- K4: layernorm(x) -> f16 for WMMA ----------------
__global__ void k_lnx(const float* x, const float* g, const float* bb, _Float16* xh) {
  int tok = blockIdx.x, f = threadIdx.x;
  __shared__ float red[128];
  float v = x[(size_t)tok * IN_ + f];
  float m  = blkReduce(v, red, f, 128) / 128.f;
  float va = blkReduce((v - m) * (v - m), red, f, 128) / 128.f;
  xh[(size_t)tok * IN_ + f] = (_Float16)((v - m) * rsqrtf(va + 1e-5f) * g[f] + bb[f]);
}

// ---------------- K5: fused projection GEMM via WMMA (4096x128 @ 128x336) ----------------
__global__ void k_proj(const _Float16* xh, const _Float16* WcatT, const float* biascat, float* proj) {
  int lane = threadIdx.x;
  int m0 = blockIdx.x * 16, n0 = blockIdx.y * 16;
  v8f acc = {};
#pragma unroll
  for (int k0 = 0; k0 < IN_; k0 += 32)
    acc = wmma_f16(ldA(xh, IN_, m0, k0, lane), ldBT(WcatT, IN_, k0, n0, lane), acc);
  int n = lane & 15, hi = lane >> 4;
#pragma unroll
  for (int i = 0; i < 8; ++i) {
    int m = m0 + i + 8 * hi, nn = n0 + n;
    proj[(size_t)m * NCOL + nn] = acc[i] + biascat[nn];
  }
}

// ---------------- K6: per-token gates / norms / regime ----------------
__global__ void k_token(const float* proj, const float* m_init,
                        const float* reg_ln_g, const float* reg_ln_b,
                        const float* reg_w2, const float* reg_b2,
                        _Float16* qh, float* qf,
                        _Float16* kn16, _Float16* knT16,
                        _Float16* vs16, _Float16* vsT16,
                        float* ofull, float* mfArr, float* fgArr,
                        float* puArr, float* sumqArr, float* igk) {
  int tok = blockIdx.x, b = tok / S_, s = tok % S_, d = threadIdx.x;
  __shared__ float red[64];
  const float* row = proj + (size_t)tok * NCOL;
  float q  = row[d];
  float k  = row[64 + d] * 0.125f;          // /sqrt(H), bias already applied pre-scale is wrong;
                                            // bias was added in GEMM then scaled here == (xW+b)/sqrt(H)
  float v  = row[128 + d];
  float ol = row[192 + d];
  float r1 = row[256 + d];
  float i_til = row[320], f_til = row[321];

  float kn2 = blkReduce(k * k, red, d, 64);
  float kn = k / fmaxf(sqrtf(kn2), 1e-12f) * 8.f;
  float vn2 = blkReduce(v * v, red, d, 64);
  float vn = v / (sqrtf(vn2) + 1e-6f) * 8.f;

  float m_prev = fminf(fmaxf(m_init[0], -100.f), 100.f);
  float m_t = fmaxf(f_til, m_prev + f_til);
  float ig = __expf(fminf(fmaxf(i_til - m_t, -15.f), 15.f));
  float fg = __expf(fminf(fmaxf(f_til + m_prev - m_t, -15.f), 15.f));
  float sc = fminf(fmaxf(ig, 0.f), 1.f);

  // regime head: gelu(LN(r1)) @ reg_w2 -> softmax -> mf = p0
  float rm = blkReduce(r1, red, d, 64) / 64.f;
  float rv = blkReduce((r1 - rm) * (r1 - rm), red, d, 64) / 64.f;
  float rn = (r1 - rm) * rsqrtf(rv + 1e-5f) * reg_ln_g[d] + reg_ln_b[d];
  float ge = 0.5f * rn * (1.f + erff(rn * 0.70710678f));
  float l0 = blkReduce(ge * reg_w2[0 * 64 + d], red, d, 64) + reg_b2[0];
  float l1 = blkReduce(ge * reg_w2[1 * 64 + d], red, d, 64) + reg_b2[1];
  float l2 = blkReduce(ge * reg_w2[2 * 64 + d], red, d, 64) + reg_b2[2];
  float l3 = blkReduce(ge * reg_w2[3 * 64 + d], red, d, 64) + reg_b2[3];
  float mx = fmaxf(fmaxf(l0, l1), fmaxf(l2, l3));
  float e0 = __expf(l0 - mx), e1 = __expf(l1 - mx), e2 = __expf(l2 - mx), e3 = __expf(l3 - mx);
  float mf = e0 / (e0 + e1 + e2 + e3);

  float mk = blkReduce(kn, red, d, 64) / 64.f;
  float mv = blkReduce(vn, red, d, 64) / 64.f;
  float sq = blkReduce(q, red, d, 64);
  float vs = sc * vn;

  size_t g = (size_t)tok * 64 + d;
  qh[g] = (_Float16)q; qf[g] = q;
  kn16[g] = (_Float16)kn; vs16[g] = (_Float16)vs;
  size_t gt = ((size_t)b * 64 + d) * S_ + s;
  knT16[gt] = (_Float16)kn; vsT16[gt] = (_Float16)vs;
  ofull[g] = 1.f / (1.f + __expf(-ol));
  igk[g] = ig * k;
  if (d == 0) { mfArr[tok] = mf; fgArr[tok] = fg; puArr[tok] = sc * mk * mv; sumqArr[tok] = sq; }
}

// ---------------- K7: C_init statistics ----------------
__global__ void k_cstats(const float* C_init, float* cstats) {
  __shared__ float r1[256], r2[256];
  int tid = threadIdx.x;
  float s = 0.f, s2 = 0.f;
  for (int i = tid; i < D_; i += 256) { float v = C_init[i]; s += v; s2 += v * v; }
  r1[tid] = s; r2[tid] = s2; __syncthreads();
  for (int off = 128; off > 0; off >>= 1) {
    if (tid < off) { r1[tid] += r1[tid + off]; r2[tid] += r2[tid + off]; } __syncthreads();
  }
  if (tid == 0) { cstats[0] = r1[0]; cstats[1] = r2[0]; }
}

// ---------------- K8: f_cum cumprod and u cumsum (u = mean_D(C_t)) ----------------
__global__ void k_fcum(const float* fg, const float* pu, const float* cstats,
                       float* fcum, float* u) {
  int b = blockIdx.x;
  if (threadIdx.x != 0) return;
  float meanCi = cstats[0] / (float)D_;
  float fc = 1.f, uc = 0.f;
  for (int s = 0; s < S_; ++s) {
    size_t tok = (size_t)b * S_ + s;
    fc *= fg[tok]; uc += pu[tok];
    fcum[tok] = fc; u[tok] = fc * meanCi + uc;
  }
}

// ---------------- K9: n_t cumsum + denominator ----------------
__global__ void k_ncum(const float* igk, const float* qf, const float* fcum,
                       const float* n_init, float* den) {
  int b = blockIdx.x, d = threadIdx.x;
  __shared__ float red[64];
  float run = 0.f; float ni = n_init[d];
  for (int s = 0; s < S_; ++s) {
    size_t tok = (size_t)b * S_ + s;
    run += igk[tok * 64 + d];
    float n = fcum[tok] * ni + run;
    float dot = blkReduce(n * qf[tok * 64 + d], red, d, 64);
    if (d == 0) den[tok] = fmaxf(fabsf(dot), 1.0f);  // THR = 1.0
  }
}

// ---------------- K10: sequential SSM scan (512 steps, DS=128 state) ----------------
__global__ void k_ssm(const float* u, const float* vol,
                      const float* log_lambda, const float* log_b, const float* ssm_c,
                      const float* log_step, const float* vol_gate,
                      const float* ln_g, const float* ln_b, float* sssm) {
  int b = blockIdx.x, n = threadIdx.x;  // n in 0..127
  __shared__ float red[128];
  float lam = -__expf(log_lambda[n]);
  float stp = __expf(log_step[n]);
  float ad  = (2.f + stp * lam) / (2.f - stp * lam);
  float bd  = stp * (1.f + ad) * __expf(log_b[n]) * 0.5f;
  float vgp = (n < 64) ? 1.f / (1.f + __expf(-vol_gate[n])) : 0.f;
  float vg  = blkReduce(vgp, red, n, 128) / 64.f;
  float h = 0.f;
  for (int t = 0; t < S_; ++t) {
    float stpt = __expf(log_step[t & (DS_ - 1)]);
    float a = (2.f + stpt * lam) / (2.f - stpt * lam);
    size_t tok = (size_t)b * S_ + t;
    h = a * h + bd * u[tok];
    float m = blkReduce(h, red, n, 128) / 128.f;
    float v = blkReduce((h - m) * (h - m), red, n, 128) / 128.f;
    h = (h - m) * rsqrtf(v + 1e-5f) * ln_g[n] + ln_b[n];
    h *= 1.f / (1.f + vg * vol[tok]);
    float so = blkReduce(h * ssm_c[n], red, n, 128);
    if (n == 0) sssm[tok] = so;
  }
}

// ---------------- K11: chunked linear-attention over C (all WMMA) ----------------
__global__ void k_chunk(const _Float16* qh, const _Float16* kn16, const _Float16* knT16,
                        const _Float16* vs16, const _Float16* vsT16,
                        const float* fcum, const float* uarrG, const float* denG,
                        const float* mfG, const float* sssmG, const float* sumqG,
                        const float* ofull, const float* C_init, const float* cstats,
                        const float* ssm_log_d, const float* ssm_alpha,
                        const float* ln_mem_g, const float* ln_mem_b,
                        float* out) {
  const int b = blockIdx.x;
  const int tid = threadIdx.x;
  const int wid = tid >> 5, lane = tid & 31;

  __shared__ _Float16 sQ[SC * 64], sK[SC * 64], sKT[64 * SC], sV[SC * 64], sVT[64 * SC];
  __shared__ _Float16 sP[SC * SC];
  __shared__ float Cb[64 * 64];       // running pure cumsum part of C
  __shared__ float Cq[SC * 64];       // (C @ q) per token in chunk
  __shared__ float Rb[SC * 64];       // scratch (R / Rinit / hn)
  __shared__ float Eb[SC * SC];       // (W W^T) .* (Kn Kn^T)
  __shared__ float tFc[SC], tU[SC], tDen[SC], tMf[SC], tS[SC], tSq[SC];
  __shared__ float rdot[SC], ridot[SC], rowE[SC], diagE[SC], sc2tot[SC];
  __shared__ float lmean[SC], linv[SC];
  __shared__ float scal[2];           // [0]=||Cb||^2 carry, [1]=<C_init,Cb> carry

  const float alpha = 1.f / (1.f + __expf(-ssm_alpha[0]));
  const float expd  = __expf(ssm_log_d[0]);
  const float g1    = alpha + (1.f - alpha) * expd;
  const float ci2   = cstats[1];

  for (int i = tid; i < 64 * 64; i += 128) Cb[i] = 0.f;
  if (tid == 0) { scal[0] = 0.f; scal[1] = 0.f; }
  __syncthreads();

  for (int c = 0; c < S_ / SC; ++c) {
    const int c0 = c * SC;
    const size_t tokBase = (size_t)b * S_ + c0;

    // S1: stage chunk operands in LDS
    for (int i = tid; i < SC * 64; i += 128) {
      int t = i >> 6, e = i & 63;
      size_t g = (tokBase + t) * 64 + e;
      sQ[i] = qh[g]; sK[i] = kn16[g]; sV[i] = vs16[g];
      size_t gt = ((size_t)b * 64 + e) * S_ + c0 + t;
      sKT[e * SC + t] = knT16[gt]; sVT[e * SC + t] = vsT16[gt];
    }
    if (c + 1 < S_ / SC && tid < 64) {  // prefetch next chunk (global_prefetch_b8)
      __builtin_prefetch(qh   + (tokBase + SC) * 64 + tid * 32, 0, 1);
      __builtin_prefetch(kn16 + (tokBase + SC) * 64 + tid * 32, 0, 1);
      __builtin_prefetch(vs16 + (tokBase + SC) * 64 + tid * 32, 0, 1);
    }
    if (tid < SC) {
      size_t g = tokBase + tid;
      tFc[tid] = fcum[g]; tU[tid] = uarrG[g]; tDen[tid] = denG[g];
      tMf[tid] = mfG[g];  tS[tid] = sssmG[g]; tSq[tid] = sumqG[g];
    }
    __syncthreads();

    // S2: P = Q Kn^T (32x32), causal mask, store f16
    {
      int m0 = (wid >> 1) * 16, n0 = (wid & 1) * 16;
      v8f acc = {};
      acc = wmma_f16(ldA(sQ, 64, m0, 0,  lane), ldBT(sK, 64, 0,  n0, lane), acc);
      acc = wmma_f16(ldA(sQ, 64, m0, 32, lane), ldBT(sK, 64, 32, n0, lane), acc);
      int n = lane & 15, hi = lane >> 4;
#pragma unroll
      for (int i = 0; i < 8; ++i) {
        int r = m0 + i + 8 * hi, cc = n0 + n;
        sP[r * SC + cc] = (_Float16)((cc <= r) ? acc[i] : 0.f);
      }
    }
    __syncthreads();

    // S3a: Rinit = Vs @ C_init ; row-dot with Kn -> ridot
#pragma unroll
    for (int r = 0; r < 2; ++r) {
      int tt = wid * 2 + r, m0 = (tt >> 2) * 16, n0 = (tt & 3) * 16;
      v8f acc = {};
      acc = wmma_f16(ldA(sV, 64, m0, 0,  lane), ldBf32(C_init, 64, 1, 0,  n0, lane), acc);
      acc = wmma_f16(ldA(sV, 64, m0, 32, lane), ldBf32(C_init, 64, 1, 32, n0, lane), acc);
      stD(Rb, 64, m0, n0, lane, acc);
    }
    __syncthreads();
    if (tid < SC) { float a = 0.f; for (int e = 0; e < 64; ++e) a += Rb[tid * 64 + e] * (float)sK[tid * 64 + e]; ridot[tid] = a; }
    __syncthreads();

    // S3b: R = Vs @ Cb ; row-dot with Kn -> rdot  (uses OLD Cb)
#pragma unroll
    for (int r = 0; r < 2; ++r) {
      int tt = wid * 2 + r, m0 = (tt >> 2) * 16, n0 = (tt & 3) * 16;
      v8f acc = {};
      acc = wmma_f16(ldA(sV, 64, m0, 0,  lane), ldBf32(Cb, 64, 1, 0,  n0, lane), acc);
      acc = wmma_f16(ldA(sV, 64, m0, 32, lane), ldBf32(Cb, 64, 1, 32, n0, lane), acc);
      stD(Rb, 64, m0, n0, lane, acc);
    }
    __syncthreads();
    if (tid < SC) { float a = 0.f; for (int e = 0; e < 64; ++e) a += Rb[tid * 64 + e] * (float)sK[tid * 64 + e]; rdot[tid] = a; }
    __syncthreads();

    // S4: Cq = Q @ Cb^T + fcum * (Q @ C_init^T)   (uses OLD Cb)
#pragma unroll
    for (int r = 0; r < 2; ++r) {
      int tt = wid * 2 + r, m0 = (tt >> 2) * 16, n0 = (tt & 3) * 16;
      v8f hb = {}, ciq = {};
      hb  = wmma_f16(ldA(sQ, 64, m0, 0,  lane), ldBf32(Cb, 1, 64, 0,  n0, lane), hb);
      hb  = wmma_f16(ldA(sQ, 64, m0, 32, lane), ldBf32(Cb, 1, 64, 32, n0, lane), hb);
      ciq = wmma_f16(ldA(sQ, 64, m0, 0,  lane), ldBf32(C_init, 1, 64, 0,  n0, lane), ciq);
      ciq = wmma_f16(ldA(sQ, 64, m0, 32, lane), ldBf32(C_init, 1, 64, 32, n0, lane), ciq);
      int n = lane & 15, hi = lane >> 4;
#pragma unroll
      for (int i = 0; i < 8; ++i) {
        int s = m0 + i + 8 * hi;
        Cq[s * 64 + n0 + n] = hb[i] + tFc[s] * ciq[i];
      }
    }

    // S5: Eb = (Vs Vs^T) .* (Kn Kn^T)
    {
      int m0 = (wid >> 1) * 16, n0 = (wid & 1) * 16;
      v8f aw = {}, ak = {};
      aw = wmma_f16(ldA(sV, 64, m0, 0,  lane), ldBT(sV, 64, 0,  n0, lane), aw);
      aw = wmma_f16(ldA(sV, 64, m0, 32, lane), ldBT(sV, 64, 32, n0, lane), aw);
      ak = wmma_f16(ldA(sK, 64, m0, 0,  lane), ldBT(sK, 64, 0,  n0, lane), ak);
      ak = wmma_f16(ldA(sK, 64, m0, 32, lane), ldBT(sK, 64, 32, n0, lane), ak);
      int n = lane & 15, hi = lane >> 4;
#pragma unroll
      for (int i = 0; i < 8; ++i) Eb[(m0 + i + 8 * hi) * SC + n0 + n] = aw[i] * ak[i];
    }
    __syncthreads();

    // S6: Cq += tril(P) @ Vs   (intra-chunk contribution)
#pragma unroll
    for (int r = 0; r < 2; ++r) {
      int tt = wid * 2 + r, m0 = (tt >> 2) * 16, n0 = (tt & 3) * 16;
      v8f acc = ldC(Cq, 64, m0, n0, lane);
      acc = wmma_f16(ldA(sP, SC, m0, 0, lane), ldBT(sVT, SC, 0, n0, lane), acc);
      stD(Cq, 64, m0, n0, lane, acc);
    }
    __syncthreads();

    // S7: Cb += Vs^T @ Kn   (chunk delta; all readers of old Cb are done)
#pragma unroll
    for (int r = 0; r < 4; ++r) {
      int tt = wid * 4 + r, m0 = (tt >> 2) * 16, n0 = (tt & 3) * 16;
      v8f acc = ldC(Cb, 64, m0, n0, lane);
      acc = wmma_f16(ldA(sVT, SC, m0, 0, lane), ldBT(sKT, SC, 0, n0, lane), acc);
      stD(Cb, 64, m0, n0, lane, acc);
    }
    __syncthreads();

    // S8: row prefixes of Eb
    if (tid < SC) {
      float a = 0.f;
      for (int t2 = 0; t2 <= tid; ++t2) a += Eb[tid * SC + t2];
      rowE[tid] = a; diagE[tid] = Eb[tid * SC + tid];
    }
    __syncthreads();

    // S9: sequential 32-step scan for ||C||_F^2 per token
    if (tid == 0) {
      float q2 = 0.f, rc = 0.f, ric = 0.f;
      float sc2p = scal[0], rib = scal[1];
      for (int t = 0; t < SC; ++t) {
        q2 += 2.f * (rowE[t] - diagE[t]) + diagE[t];
        rc += rdot[t]; ric += ridot[t];
        float sc2p_t = sc2p + 2.f * rc + q2;
        float fc = tFc[t];
        sc2tot[t] = fc * fc * ci2 + 2.f * fc * (rib + ric) + sc2p_t;
      }
      scal[0] = sc2p + 2.f * rc + q2;
      scal[1] = rib + ric;
    }
    __syncthreads();

    // S10: per-token affine mix + frob norm + h_num/den, then LN + output gate
    for (int i = tid; i < SC * 64; i += 128) {
      int s = i >> 6, d = i & 63;
      float A  = 1.f + tMf[s] * (g1 - 1.f);
      float Bc = tMf[s] * (1.f - alpha) * tS[s];
      float S1 = (float)D_ * tU[s];
      float fr2 = A * A * sc2tot[s] + 2.f * A * Bc * S1 + Bc * Bc * (float)D_;
      float frob = sqrtf(fmaxf(fr2, 0.f));
      float val = (A * Cq[s * 64 + d] + Bc * tSq[s]) * 8.f / ((frob + 1e-6f) * tDen[s]);
      Rb[i] = val;
    }
    __syncthreads();
    if (tid < SC) {
      float m = 0.f; for (int d = 0; d < 64; ++d) m += Rb[tid * 64 + d]; m /= 64.f;
      float v = 0.f; for (int d = 0; d < 64; ++d) { float z = Rb[tid * 64 + d] - m; v += z * z; } v /= 64.f;
      lmean[tid] = m; linv[tid] = rsqrtf(v + 1e-5f);
    }
    __syncthreads();
    for (int i = tid; i < SC * 64; i += 128) {
      int s = i >> 6, d = i & 63;
      size_t g = (tokBase + s) * 64 + d;
      float y = (Rb[i] - lmean[s]) * linv[s] * ln_mem_g[d] + ln_mem_b[d];
      out[g] = ofull[g] * y;
    }
    __syncthreads();
  }
}

// ---------------- host launch ----------------
extern "C" void kernel_launch(void* const* d_in, const int* in_sizes, int n_in,
                              void* d_out, int out_size, void* d_ws, size_t ws_size,
                              hipStream_t stream) {
  const float* x      = (const float*)d_in[0];
  const float* Wq_w   = (const float*)d_in[1];  const float* Wq_b = (const float*)d_in[2];
  const float* Wk_w   = (const float*)d_in[3];  const float* Wk_b = (const float*)d_in[4];
  const float* Wv_w   = (const float*)d_in[5];  const float* Wv_b = (const float*)d_in[6];
  const float* Wi_w   = (const float*)d_in[7];  const float* Wi_b = (const float*)d_in[8];
  const float* Wf_w   = (const float*)d_in[9];  const float* Wf_b = (const float*)d_in[10];
  const float* Wo_w   = (const float*)d_in[11]; const float* Wo_b = (const float*)d_in[12];
  const float* ln_in_g = (const float*)d_in[13]; const float* ln_in_b = (const float*)d_in[14];
  const float* ln_mem_g = (const float*)d_in[15]; const float* ln_mem_b = (const float*)d_in[16];
  const float* reg_w1 = (const float*)d_in[17]; const float* reg_b1 = (const float*)d_in[18];
  const float* reg_ln_g = (const float*)d_in[19]; const float* reg_ln_b = (const float*)d_in[20];
  const float* reg_w2 = (const float*)d_in[21]; const float* reg_b2 = (const float*)d_in[22];
  const float* C_init = (const float*)d_in[23];
  const float* n_init = (const float*)d_in[24];
  const float* m_init = (const float*)d_in[25];
  const float* ssm_log_lambda = (const float*)d_in[26];
  const float* ssm_log_b = (const float*)d_in[27];
  const float* ssm_c = (const float*)d_in[28];
  const float* ssm_log_d = (const float*)d_in[29];
  const float* ssm_log_step = (const float*)d_in[30];
  const float* ssm_vol_gate = (const float*)d_in[31];
  const float* ssm_alpha = (const float*)d_in[32];
  const float* ssm_ln_g = (const float*)d_in[33];
  const float* ssm_ln_b = (const float*)d_in[34];
  float* out = (float*)d_out;

  char* w = (char*)d_ws;
  auto alloc = [&](size_t bytes) -> void* {
    void* p = (void*)w; w += (bytes + 255) & ~(size_t)255; return p;
  };
  const size_t BS = (size_t)B_ * S_;
  float*     cs      = (float*)alloc((size_t)B_ * (S_ + 1) * IN_ * 4);
  float*     cs2     = (float*)alloc((size_t)B_ * (S_ + 1) * IN_ * 4);
  float*     volraw  = (float*)alloc(BS * 4);
  float*     vol     = (float*)alloc(BS * 4);
  _Float16*  xh      = (_Float16*)alloc(BS * IN_ * 2);
  _Float16*  WcatT   = (_Float16*)alloc((size_t)NCOL * IN_ * 2);
  float*     biascat = (float*)alloc(NCOL * 4);
  float*     proj    = (float*)alloc(BS * NCOL * 4);
  _Float16*  qh      = (_Float16*)alloc(BS * 64 * 2);
  float*     qf      = (float*)alloc(BS * 64 * 4);
  _Float16*  kn16    = (_Float16*)alloc(BS * 64 * 2);
  _Float16*  knT16   = (_Float16*)alloc(BS * 64 * 2);
  _Float16*  vs16    = (_Float16*)alloc(BS * 64 * 2);
  _Float16*  vsT16   = (_Float16*)alloc(BS * 64 * 2);
  float*     ofull   = (float*)alloc(BS * 64 * 4);
  float*     mfArr   = (float*)alloc(BS * 4);
  float*     fgArr   = (float*)alloc(BS * 4);
  float*     puArr   = (float*)alloc(BS * 4);
  float*     sumqArr = (float*)alloc(BS * 4);
  float*     igk     = (float*)alloc(BS * 64 * 4);
  float*     fcum    = (float*)alloc(BS * 4);
  float*     uarr    = (float*)alloc(BS * 4);
  float*     den     = (float*)alloc(BS * 4);
  float*     sssm    = (float*)alloc(BS * 4);
  float*     cstats  = (float*)alloc(2 * 4);

  k_prep<<<NCOL, IN_, 0, stream>>>(Wq_w, Wq_b, Wk_w, Wk_b, Wv_w, Wv_b, Wi_w, Wi_b,
                                   Wf_w, Wf_b, Wo_w, Wo_b, reg_w1, reg_b1, WcatT, biascat);
  k_cumsum<<<B_, IN_, 0, stream>>>(x, cs, cs2);
  k_vol<<<B_ * S_, IN_, 0, stream>>>(cs, cs2, volraw);
  k_volnorm<<<B_, 256, 0, stream>>>(volraw, vol);
  k_lnx<<<B_ * S_, IN_, 0, stream>>>(x, ln_in_g, ln_in_b, xh);
  k_proj<<<dim3(256, NCOL / 16), 32, 0, stream>>>(xh, WcatT, biascat, proj);
  k_token<<<B_ * S_, 64, 0, stream>>>(proj, m_init, reg_ln_g, reg_ln_b, reg_w2, reg_b2,
                                      qh, qf, kn16, knT16, vs16, vsT16,
                                      ofull, mfArr, fgArr, puArr, sumqArr, igk);
  k_cstats<<<1, 256, 0, stream>>>(C_init, cstats);
  k_fcum<<<B_, 32, 0, stream>>>(fgArr, puArr, cstats, fcum, uarr);
  k_ncum<<<B_, 64, 0, stream>>>(igk, qf, fcum, n_init, den);
  k_ssm<<<B_, 128, 0, stream>>>(uarr, vol, ssm_log_lambda, ssm_log_b, ssm_c,
                                ssm_log_step, ssm_vol_gate, ssm_ln_g, ssm_ln_b, sssm);
  k_chunk<<<B_, 128, 0, stream>>>(qh, kn16, knT16, vs16, vsT16,
                                  fcum, uarr, den, mfArr, sssm, sumqArr,
                                  ofull, C_init, cstats, ssm_log_d, ssm_alpha,
                                  ln_mem_g, ln_mem_b, out);
}
